// MemoryRetriever_77068893160269
// MI455X (gfx1250) — matrix operand order repaired and checked by reading, not compiled, not run on hardware
//
#include <hip/hip_runtime.h>
#include <hip/hip_bf16.h>

#define B_   2048
#define M_   2048
#define H_   64
#define D_   64
#define V_   128
#define G_   32

typedef __attribute__((ext_vector_type(16))) _Float16 v16h;
typedef __attribute__((ext_vector_type(8)))  float    v8f;

// ---- layout helpers (wave32); A/B use this code's contiguous K order, not ISA 7.12.2's ------------------------------------
// A matrix (16x32 f16): lane halves split K. For lane l, element j of v16h:
//   row M = l & 15 ;  K = (l>>4)*16 + j
__device__ __forceinline__ int a_k_idx(int lane, int j) {
    return (lane >> 4) * 16 + j;
}
// B matrix (32x16 f16): N = lane & 15 ; K = (lane>>4)*16 + j
// C/D (16x16 f32, 8 VGPRs): M = r + 8*(lane>>4) ; N = lane & 15

// ---- Kernel 1: q_proj = query @ Wq + bq  (WMMA) ----------------------------
__global__ __launch_bounds__(256)
void k_qproj(const float* __restrict__ query, const float* __restrict__ Wq,
             const float* __restrict__ bq, float* __restrict__ qp) {
    int wave = (blockIdx.x * blockDim.x + threadIdx.x) >> 5;   // 0..511
    int lane = threadIdx.x & 31;
    int tn = wave & 3;            // D tile (4)
    int tm = wave >> 2;           // B tile (128)
    int arow = tm * 16 + (lane & 15);
    int ncol = tn * 16 + (lane & 15);
    v8f c = {};
#pragma unroll
    for (int kc = 0; kc < 2; ++kc) {
        int k0 = kc * 32;
        v16h a, b;
#pragma unroll
        for (int j = 0; j < 16; ++j) {
            a[j] = (_Float16)query[arow * H_ + k0 + a_k_idx(lane, j)];
            b[j] = (_Float16)Wq[(k0 + (lane >> 4) * 16 + j) * D_ + ncol];
        }
        c = __builtin_amdgcn_wmma_f32_16x16x32_f16(false, a, false, b,
                                                   (short)0, c, false, false);
    }
    float bias = bq[ncol];
#pragma unroll
    for (int r = 0; r < 8; ++r) {
        int m = tm * 16 + r + 8 * (lane >> 4);
        qp[m * D_ + ncol] = c[r] + bias;
    }
}

// ---- Kernel 2: fused sims -> online softmax -> retrieval (one HBM pass) ----
// 1 block per batch, 8 waves. Each wave processes 2 rows/iter with b128 loads:
// lanes 0-15 handle row m, lanes 16-31 row m+1; each half-wave keeps its own
// online-softmax state (mx, s, 64-wide accumulator as float4/lane).
__global__ __launch_bounds__(256)
void k_retrieve(const float* __restrict__ qp, const float* __restrict__ memory,
                float* __restrict__ retrieved, float* __restrict__ max_sim) {
    const int NP = 16;                      // 8 waves x 2 half-waves
    int b    = blockIdx.x;
    int wave = threadIdx.x >> 5;
    int lane = threadIdx.x & 31;
    int half = lane >> 4;                   // which row of the pair
    int q    = lane & 15;                   // float4 slot within row (D=64)

    const float4* mem4 = (const float4*)(memory + (size_t)b * M_ * D_);
    float4 q4 = ((const float4*)(qp + (size_t)b * D_))[q];

    float  mx = -INFINITY, s = 0.f;
    float4 acc; acc.x = acc.y = acc.z = acc.w = 0.f;

    const int rows = M_ / 8;                // 256 rows per wave
    const int m0   = wave * rows;
#pragma unroll 4
    for (int m = m0; m < m0 + rows; m += 2) {
        // speculative stream-ahead prefetch (global_prefetch_b8), 32 rows ahead
        __builtin_prefetch(&mem4[(size_t)(m + 32 + half) * 16 + q], 0, 0);

        float4 v = mem4[(size_t)(m + half) * 16 + q];     // b128, coalesced
        float p = v.x * q4.x + v.y * q4.y + v.z * q4.z + v.w * q4.w;
        p += __shfl_xor(p, 8);                            // reduce within half
        p += __shfl_xor(p, 4);
        p += __shfl_xor(p, 2);
        p += __shfl_xor(p, 1);
        p *= 0.125f;                                      // 1/sqrt(D), D=64
        float nm   = fmaxf(mx, p);
        float corr = __expf(mx - nm);                     // 0 on first iter
        float e    = __expf(p - nm);
        s = s * corr + e;
        acc.x = fmaf(e, v.x, acc.x * corr);
        acc.y = fmaf(e, v.y, acc.y * corr);
        acc.z = fmaf(e, v.z, acc.z * corr);
        acc.w = fmaf(e, v.w, acc.w * corr);
        mx = nm;
    }

    __shared__ float smx[NP], ss[NP];
    __shared__ float sacc[NP][D_];
    int part = wave * 2 + half;
    sacc[part][4 * q + 0] = acc.x;
    sacc[part][4 * q + 1] = acc.y;
    sacc[part][4 * q + 2] = acc.z;
    sacc[part][4 * q + 3] = acc.w;
    if (q == 0) { smx[part] = mx; ss[part] = s; }
    __syncthreads();

    if (threadIdx.x < D_) {
        int d = threadIdx.x;
        float gm = smx[0];
#pragma unroll
        for (int w = 1; w < NP; ++w) gm = fmaxf(gm, smx[w]);
        float st = 0.f, ad = 0.f;
#pragma unroll
        for (int w = 0; w < NP; ++w) {
            float f = __expf(smx[w] - gm);
            st += ss[w] * f;
            ad += sacc[w][d] * f;
        }
        retrieved[b * D_ + d] = ad / st;
        if (d == 0) max_sim[b] = gm;
    }
}

// ---- Kernel 3: gate MLP ----------------------------------------------------
__global__ __launch_bounds__(256)
void k_gate(const float* __restrict__ query, const float* __restrict__ max_sim,
            const float* __restrict__ Wg1, const float* __restrict__ bg1,
            const float* __restrict__ Wg2, const float* __restrict__ bg2,
            float* __restrict__ gate_out) {
    int b = blockIdx.x * blockDim.x + threadIdx.x;
    if (b >= B_) return;
    float accs[G_];
#pragma unroll
    for (int g = 0; g < G_; ++g) accs[g] = bg1[g];
    for (int h = 0; h < H_; ++h) {
        float qh = query[b * H_ + h];
#pragma unroll
        for (int g = 0; g < G_; ++g)
            accs[g] = fmaf(qh, Wg1[h * G_ + g], accs[g]);
    }
    float ms = max_sim[b];
#pragma unroll
    for (int g = 0; g < G_; ++g)
        accs[g] = fmaf(ms, Wg1[H_ * G_ + g], accs[g]);
    float z = bg2[0];
#pragma unroll
    for (int g = 0; g < G_; ++g)
        z = fmaf(fmaxf(accs[g], 0.f), Wg2[g], z);
    gate_out[b] = 1.f / (1.f + __expf(-z));
}

// ---- Kernel 4: retrieved@Wo+bo -> gate blend -> @Wc+bc (WMMA) --------------
__global__ __launch_bounds__(256)
void k_output(const float* __restrict__ retrieved, const float* __restrict__ gate,
              const float* __restrict__ Wo, const float* __restrict__ bo,
              const float* __restrict__ null_vec,
              const float* __restrict__ Wc, const float* __restrict__ bc,
              float* __restrict__ logits) {
    __shared__ _Float16 Ash[16 * D_];   // retrieved tile (f16)
    __shared__ _Float16 Osh[16 * H_];   // blended out tile (f16)
    __shared__ float    gsh[16];
    __shared__ float    nsh[H_];

    int b0  = blockIdx.x * 16;
    int tid = threadIdx.x;
    for (int i = tid; i < 16 * D_; i += 256)
        Ash[i] = (_Float16)retrieved[(b0 + (i >> 6)) * D_ + (i & 63)];
    if (tid < 16) gsh[tid] = gate[b0 + tid];
    if (tid < H_) nsh[tid] = null_vec[tid];
    __syncthreads();

    int wave = tid >> 5, lane = tid & 31;

    // GEMM1: rh = retrieved @ Wo + bo ; out = g*rh + (1-g)*null  (4 waves)
    if (wave < 4) {
        int n = wave * 16 + (lane & 15);
        v8f c = {};
#pragma unroll
        for (int kc = 0; kc < 2; ++kc) {
            int k0 = kc * 32;
            v16h a, b;
#pragma unroll
            for (int j = 0; j < 16; ++j) {
                a[j] = Ash[(lane & 15) * D_ + k0 + a_k_idx(lane, j)];
                b[j] = (_Float16)Wo[(k0 + (lane >> 4) * 16 + j) * H_ + n];
            }
            c = __builtin_amdgcn_wmma_f32_16x16x32_f16(false, a, false, b,
                                                       (short)0, c, false, false);
        }
        float bias = bo[n];
#pragma unroll
        for (int r = 0; r < 8; ++r) {
            int i  = r + 8 * (lane >> 4);
            float g  = gsh[i];
            float o  = g * (c[r] + bias) + (1.f - g) * nsh[n];
            Osh[i * H_ + n] = (_Float16)o;
        }
    }
    __syncthreads();

    // GEMM2: logits = out @ Wc + bc  (8 waves, V=128)
    {
        int n = wave * 16 + (lane & 15);
        v8f c = {};
#pragma unroll
        for (int kc = 0; kc < 2; ++kc) {
            int k0 = kc * 32;
            v16h a, b;
#pragma unroll
            for (int j = 0; j < 16; ++j) {
                a[j] = Osh[(lane & 15) * H_ + k0 + a_k_idx(lane, j)];
                b[j] = (_Float16)Wc[(k0 + (lane >> 4) * 16 + j) * V_ + n];
            }
            c = __builtin_amdgcn_wmma_f32_16x16x32_f16(false, a, false, b,
                                                       (short)0, c, false, false);
        }
        float bias = bc[n];
#pragma unroll
        for (int r = 0; r < 8; ++r) {
            int i = r + 8 * (lane >> 4);
            logits[(size_t)(b0 + i) * V_ + n] = c[r] + bias;
        }
    }
}

extern "C" void kernel_launch(void* const* d_in, const int* in_sizes, int n_in,
                              void* d_out, int out_size, void* d_ws, size_t ws_size,
                              hipStream_t stream) {
    const float* query    = (const float*)d_in[0];
    const float* memory   = (const float*)d_in[1];
    const float* Wq       = (const float*)d_in[2];
    const float* bq       = (const float*)d_in[3];
    const float* Wo       = (const float*)d_in[4];
    const float* bo       = (const float*)d_in[5];
    const float* Wg1      = (const float*)d_in[6];
    const float* bg1      = (const float*)d_in[7];
    const float* Wg2      = (const float*)d_in[8];
    const float* bg2      = (const float*)d_in[9];
    const float* null_vec = (const float*)d_in[10];
    const float* Wc       = (const float*)d_in[11];
    const float* bc       = (const float*)d_in[12];

    float* logits = (float*)d_out;                     // (B, V)
    float* gate   = (float*)d_out + (size_t)B_ * V_;   // (B,)

    float* ws        = (float*)d_ws;
    float* qp        = ws;                       // B*D
    float* retrieved = ws + (size_t)B_ * D_;     // B*D
    float* max_sim   = ws + (size_t)2 * B_ * D_; // B

    // 1) q_proj (WMMA): 512 wave-tiles -> 64 blocks x 256
    k_qproj<<<64, 256, 0, stream>>>(query, Wq, bq, qp);
    // 2) fused one-pass retrieval: one block per batch
    k_retrieve<<<B_, 256, 0, stream>>>(qp, memory, retrieved, max_sim);
    // 3) gate MLP
    k_gate<<<B_ / 256, 256, 0, stream>>>(query, max_sim, Wg1, bg1, Wg2, bg2, gate);
    // 4) output GEMMs (WMMA): 16 batches per block
    k_output<<<B_ / 16, 256, 0, stream>>>(retrieved, gate, Wo, bo, null_vec, Wc, bc, logits);
}
